// BiGCN_52716428591487
// MI455X (gfx1250) — matrix operand order repaired and checked by reading, not compiled, not run on hardware
//
#include <hip/hip_runtime.h>
#include <stdint.h>

// ---------------- problem constants (fixed by the reference) ----------------
#define NN     50000
#define GG     100
#define NPG    500
#define EE     800000
#define FIN    256
#define HIDDEN 128
#define NC     256   // output cols of both fused GEMMs

typedef _Float16 f16;
typedef _Float16 v8h  __attribute__((ext_vector_type(8)));
typedef _Float16 v16h __attribute__((ext_vector_type(16)));
typedef float    v8f  __attribute__((ext_vector_type(8)));

// ---------------- elementwise / prep kernels ----------------

__global__ void k_cvt_x(const float* __restrict__ x, f16* __restrict__ xh, int n) {
    int i = blockIdx.x * blockDim.x + threadIdx.x;
    if (i < n) xh[i] = (f16)x[i];
}

// W1t[n][k] = concat(W_bu1, W_td1) column n, transposed to row-major-by-column
__global__ void k_prep_w1(const float* __restrict__ Wbu, const float* __restrict__ Wtd,
                          f16* __restrict__ Wt) {
    int i = blockIdx.x * blockDim.x + threadIdx.x;   // 256*256
    int nn = i >> 8, k = i & 255;
    float v = (nn < HIDDEN) ? Wbu[k * HIDDEN + nn] : Wtd[k * HIDDEN + (nn - HIDDEN)];
    Wt[i] = (f16)v;
}

// Block-diagonal layer-2 weight (bottom 128 rows of W_bu2 / W_td2), transposed.
__global__ void k_prep_w2(const float* __restrict__ Wbu2, const float* __restrict__ Wtd2,
                          f16* __restrict__ Wt) {
    int i = blockIdx.x * blockDim.x + threadIdx.x;   // 256*256
    int nn = i >> 8, k = i & 255;
    float v = 0.f;
    if (nn < HIDDEN) { if (k <  HIDDEN) v = Wbu2[(FIN + k)          * HIDDEN + nn]; }
    else             { if (k >= HIDDEN) v = Wtd2[(FIN + (k-HIDDEN)) * HIDDEN + (nn-HIDDEN)]; }
    Wt[i] = (f16)v;
}

__global__ void k_deg_init(float* __restrict__ a, float* __restrict__ b, int n) {
    int i = blockIdx.x * blockDim.x + threadIdx.x;
    if (i < n) { a[i] = 1.f; b[i] = 1.f; }
}

__global__ void k_deg_edges(const int* __restrict__ src, const int* __restrict__ dst,
                            float* __restrict__ deg_bu, float* __restrict__ deg_td, int e) {
    int i = blockIdx.x * blockDim.x + threadIdx.x;
    if (i < e) {
        atomicAdd(&deg_bu[src[i]], 1.f);   // bu direction: degree counted at original src
        atomicAdd(&deg_td[dst[i]], 1.f);   // td direction: degree counted at original dst
    }
}

__global__ void k_rsqrt2(float* __restrict__ a, float* __restrict__ b, int n) {
    int i = blockIdx.x * blockDim.x + threadIdx.x;
    if (i < n) { a[i] = rsqrtf(a[i]); b[i] = rsqrtf(b[i]); }
}

// ---------------- WMMA GEMM: C[M,256] = A[M,K](f16) @ Bt[256,K]^T (+ per-graph RW) --------
// A fragment layout (16-bit A 16x32, ISA 7.12.2):
//   lane l<16 : row M=l,    halves 0..7 = K k0..k0+7,  halves 8..15 = K k0+16..k0+23 (k0 = kb)
//   lane l>=16: row M=l-16, k0 = kb+8
__device__ __forceinline__ v16h load_a_frag(const f16* __restrict__ base, int ld,
                                            int row_base, int kb, int lane) {
    int r  = row_base + (lane & 15);
    int k0 = kb + ((lane >> 4) << 3);
    const f16* p = base + (size_t)r * ld + k0;
    v8h lo = *(const v8h*)(p);
    v8h hi = *(const v8h*)(p + 16);
    v16h f;
#pragma unroll
    for (int i = 0; i < 8; ++i) { f[i] = lo[i]; f[i + 8] = hi[i]; }
    return f;
}

// B fragment (16-bit B 32x16): lane l holds column N=l&15; halves 0..7 = K k0..k0+7,
// halves 8..15 = K k0+16..k0+23, k0 = kb (+8 for lanes>=16). Bt is [Ncols][K] row-major.
__device__ __forceinline__ v16h load_b_frag(const f16* __restrict__ bt, int ld,
                                            int col_base, int kb, int lane) {
    int c  = col_base + (lane & 15);
    int k0 = kb + ((lane >> 4) << 3);
    const f16* p = bt + (size_t)c * ld + k0;
    v8h lo = *(const v8h*)(p);
    v8h hi = *(const v8h*)(p + 16);
    v16h f;
#pragma unroll
    for (int i = 0; i < 8; ++i) { f[i] = lo[i]; f[i + 8] = hi[i]; }
    return f;
}

// 128 threads = 4 waves; wave w owns cols [w*64, w*64+64) as 4 N-tiles (A-frag reuse x4).
// One block = one 16-row stripe across all 256 output columns. M % 16 == 0 (50000/16=3125).
__global__ __launch_bounds__(128)
void k_wmma_gemm(const f16* __restrict__ A, const f16* __restrict__ Bt,
                 float* __restrict__ C, int K,
                 const float* __restrict__ rw, int npg) {
    int lane = threadIdx.x & 31;
    int wave = threadIdx.x >> 5;
    int row_base = blockIdx.x << 4;
    int col0 = wave << 6;

    v8f acc[4] = {};
    for (int kb = 0; kb < K; kb += 32) {
        v16h a = load_a_frag(A, K, row_base, kb, lane);
#pragma unroll
        for (int t = 0; t < 4; ++t) {
            v16h b = load_b_frag(Bt, K, col0 + (t << 4), kb, lane);
            acc[t] = __builtin_amdgcn_wmma_f32_16x16x32_f16(
                false, a, false, b, (short)0, acc[t], false, false);
        }
    }
    // C/D layout: VGPR r -> row (r | r+8 for lanes>=16), col = lane&15
    int rlo = row_base + ((lane >> 4) << 3);
    int cw  = lane & 15;
#pragma unroll
    for (int t = 0; t < 4; ++t) {
        int col = col0 + (t << 4) + cw;
#pragma unroll
        for (int r = 0; r < 8; ++r) {
            int row = rlo + r;
            float v = acc[t][r];
            if (rw) v += rw[(row / npg) * NC + col];   // broadcast per-graph root term
            C[(size_t)row * NC + col] = v;
        }
    }
}

// ---------------- layer-1 aggregation ----------------

// A1[i][c] = H1[i][c] * dinv^2 (self-loop term); cols 0..127 bu, 128..255 td
__global__ void k_a1_init(const float* __restrict__ H1, const float* __restrict__ dbu,
                          const float* __restrict__ dtd, float* __restrict__ A1, int total) {
    int i = blockIdx.x * blockDim.x + threadIdx.x;
    if (i >= total) return;
    int node = i >> 8, c = i & 255;
    float dv = (c < HIDDEN) ? dbu[node] : dtd[node];
    A1[i] = H1[i] * dv * dv;
}

// warp per edge: bu aggregates at src from dst; td aggregates at dst from src
__global__ void k_scatter1(const int* __restrict__ src, const int* __restrict__ dst,
                           const float* __restrict__ H1, const float* __restrict__ dbu,
                           const float* __restrict__ dtd, float* __restrict__ A1, int e) {
    int gtid = blockIdx.x * blockDim.x + threadIdx.x;
    int warp = gtid >> 5, lane = gtid & 31;
    int nwarps = (gridDim.x * blockDim.x) >> 5;
    for (int ed = warp; ed < e; ed += nwarps) {
        int s = src[ed], d = dst[ed];
        float cbu = dbu[s] * dbu[d];
        float ctd = dtd[s] * dtd[d];
        float4 hb = *(const float4*)(H1 + (size_t)d * 256 + lane * 4);        // bu half
        float* abu = A1 + (size_t)s * 256 + lane * 4;
        atomicAdd(abu + 0, hb.x * cbu); atomicAdd(abu + 1, hb.y * cbu);
        atomicAdd(abu + 2, hb.z * cbu); atomicAdd(abu + 3, hb.w * cbu);
        float4 ht = *(const float4*)(H1 + (size_t)s * 256 + 128 + lane * 4);  // td half
        float* atd = A1 + (size_t)d * 256 + 128 + lane * 4;
        atomicAdd(atd + 0, ht.x * ctd); atomicAdd(atd + 1, ht.y * ctd);
        atomicAdd(atd + 2, ht.z * ctd); atomicAdd(atd + 3, ht.w * ctd);
    }
}

// Z[i][c] = relu(A1[i][c] + b1[c]) as f16
__global__ void k_build_z(const float* __restrict__ A1, const float* __restrict__ bbu1,
                          const float* __restrict__ btd1, f16* __restrict__ Z, int total) {
    int i = blockIdx.x * blockDim.x + threadIdx.x;
    if (i >= total) return;
    int c = i & 255;
    float b = (c < HIDDEN) ? bbu1[c] : btd1[c - HIDDEN];
    Z[i] = (f16)fmaxf(A1[i] + b, 0.f);
}

// RW[g][n] = relu(x[root_g]) @ top-256-rows of W_bu2 / W_td2 (per-graph constant)
__global__ void k_rw(const float* __restrict__ x, const float* __restrict__ Wbu2,
                     const float* __restrict__ Wtd2, float* __restrict__ RW) {
    int i = blockIdx.x * blockDim.x + threadIdx.x;
    if (i >= GG * NC) return;
    int g = i >> 8, nn = i & 255;
    const float* xr = x + (size_t)(g * NPG) * FIN;
    const float* w  = (nn < HIDDEN) ? (Wbu2 + nn) : (Wtd2 + (nn - HIDDEN));
    float sum = 0.f;
    for (int k = 0; k < FIN; ++k) sum += fmaxf(xr[k], 0.f) * w[(size_t)k * HIDDEN];
    RW[i] = sum;
}

// ---------------- layer-2 aggregation (roots only) ----------------

__global__ void k_out2_init(const float* __restrict__ H2, const float* __restrict__ dtd,
                            const float* __restrict__ bbu2, const float* __restrict__ btd2,
                            float* __restrict__ o2) {
    int i = blockIdx.x * blockDim.x + threadIdx.x;
    if (i >= GG * NC) return;
    int g = i >> 8, c = i & 255;
    int r = g * NPG;
    float dv = dtd[r];
    float b = (c < HIDDEN) ? bbu2[c] : btd2[c - HIDDEN];
    o2[i] = H2[(size_t)r * NC + c] * dv * dv + b;
}

__global__ void k_scatter2(const int* __restrict__ src, const int* __restrict__ dst,
                           const float* __restrict__ H2, const float* __restrict__ dtd,
                           float* __restrict__ o2, int e) {
    int gtid = blockIdx.x * blockDim.x + threadIdx.x;
    int warp = gtid >> 5, lane = gtid & 31;
    int nwarps = (gridDim.x * blockDim.x) >> 5;
    for (int ed = warp; ed < e; ed += nwarps) {
        int d = dst[ed];
        if (d % NPG != 0) continue;            // only edges landing on a root matter
        int g = d / NPG;
        int s = src[ed];
        float coef = dtd[s] * dtd[d];
        const float* h = H2 + (size_t)s * NC + lane * 8;
        float* o = o2 + (size_t)g * NC + lane * 8;
#pragma unroll
        for (int j = 0; j < 8; ++j) atomicAdd(o + j, h[j] * coef);
    }
}

// relu -> [256]@[256,4] -> log_softmax, one block per graph
__global__ void k_head(const float* __restrict__ o2, const float* __restrict__ Wlin,
                       const float* __restrict__ blin, float* __restrict__ out) {
    __shared__ float sl[4];
    int g = blockIdx.x;
    int t = threadIdx.x;                       // 256 threads
    if (t < 4) sl[t] = blin[t];
    __syncthreads();
    float v = fmaxf(o2[(size_t)g * NC + t], 0.f);
#pragma unroll
    for (int j = 0; j < 4; ++j) atomicAdd(&sl[j], v * Wlin[t * 4 + j]);
    __syncthreads();
    if (t == 0) {
        float m = fmaxf(fmaxf(sl[0], sl[1]), fmaxf(sl[2], sl[3]));
        float s = 0.f;
        for (int j = 0; j < 4; ++j) s += expf(sl[j] - m);
        float lse = logf(s);
        for (int j = 0; j < 4; ++j) out[g * 4 + j] = sl[j] - m - lse;
    }
}

// ---------------- host launcher ----------------

extern "C" void kernel_launch(void* const* d_in, const int* in_sizes, int n_in,
                              void* d_out, int out_size, void* d_ws, size_t ws_size,
                              hipStream_t stream) {
    const float* x    = (const float*)d_in[0];
    const int*   ei   = (const int*)d_in[1];
    const int*   src  = ei;
    const int*   dst  = ei + EE;
    const float* Wbu1 = (const float*)d_in[4];
    const float* bbu1 = (const float*)d_in[5];
    const float* Wtd1 = (const float*)d_in[6];
    const float* btd1 = (const float*)d_in[7];
    const float* Wbu2 = (const float*)d_in[8];
    const float* bbu2 = (const float*)d_in[9];
    const float* Wtd2 = (const float*)d_in[10];
    const float* btd2 = (const float*)d_in[11];
    const float* Wlin = (const float*)d_in[12];
    const float* blin = (const float*)d_in[13];
    float* out = (float*)d_out;
    (void)in_sizes; (void)n_in; (void)out_size; (void)ws_size;

    char* ws = (char*)d_ws;
    size_t off = 0;
    auto alloc = [&](size_t bytes) -> char* {
        char* p = ws + off;
        off += (bytes + 255) & ~(size_t)255;
        return p;
    };
    f16*   xh  = (f16*)alloc((size_t)NN * FIN * 2);   // x in f16
    f16*   W1t = (f16*)alloc((size_t)256 * 256 * 2);  // [Wbu1|Wtd1]^T
    f16*   W2t = (f16*)alloc((size_t)256 * 256 * 2);  // blockdiag layer-2 weight^T
    f16*   Z   = (f16*)alloc((size_t)NN * NC * 2);    // relu(l1+bias) in f16
    float* H1  = (float*)alloc((size_t)NN * NC * 4);  // GEMM1 out; reused as GEMM2 out (H2)
    float* A1  = (float*)alloc((size_t)NN * NC * 4);  // layer-1 aggregate
    float* dbu = (float*)alloc((size_t)NN * 4);
    float* dtd = (float*)alloc((size_t)NN * 4);
    float* RW  = (float*)alloc((size_t)GG * NC * 4);
    float* o2  = (float*)alloc((size_t)GG * NC * 4);

    const int T = 256;
    k_cvt_x   <<<(NN * FIN + T - 1) / T, T, 0, stream>>>(x, xh, NN * FIN);
    k_prep_w1 <<<(256 * 256) / T, T, 0, stream>>>(Wbu1, Wtd1, W1t);
    k_prep_w2 <<<(256 * 256) / T, T, 0, stream>>>(Wbu2, Wtd2, W2t);
    k_deg_init<<<(NN + T - 1) / T, T, 0, stream>>>(dbu, dtd, NN);
    k_deg_edges<<<(EE + T - 1) / T, T, 0, stream>>>(src, dst, dbu, dtd, EE);
    k_rsqrt2  <<<(NN + T - 1) / T, T, 0, stream>>>(dbu, dtd, NN);

    // layer 1: H1 = xh @ [Wbu1|Wtd1]
    k_wmma_gemm<<<NN / 16, 128, 0, stream>>>(xh, W1t, H1, FIN, nullptr, NPG);
    k_a1_init <<<(NN * NC + T - 1) / T, T, 0, stream>>>(H1, dbu, dtd, A1, NN * NC);
    k_scatter1<<<1024, 256, 0, stream>>>(src, dst, H1, dbu, dtd, A1, EE);
    k_build_z <<<(NN * NC + T - 1) / T, T, 0, stream>>>(A1, bbu1, btd1, Z, NN * NC);
    k_rw      <<<(GG * NC + T - 1) / T, T, 0, stream>>>(x, Wbu2, Wtd2, RW);

    // layer 2: H2 = Z @ blockdiag + RW[graph]  (written over H1 buffer)
    k_wmma_gemm<<<NN / 16, 128, 0, stream>>>(Z, W2t, H1, NC, RW, NPG);
    k_out2_init<<<(GG * NC + T - 1) / T, T, 0, stream>>>(H1, dtd, bbu2, btd2, o2);
    k_scatter2 <<<1024, 256, 0, stream>>>(src, dst, H1, dtd, o2, EE);
    k_head     <<<GG, 256, 0, stream>>>(o2, Wlin, blin, out);
}
